// Model1_23922967839342
// MI455X (gfx1250) — matrix operand order repaired and checked by reading, not compiled, not run on hardware
//
#include <hip/hip_runtime.h>
#include <hip/hip_bf16.h>

// Model dims
#define BDIM 512
#define TDIM 64
#define NDIM 32
#define MDIM 512
#define CDIM 64
#define KHDIM 5
#define TMDIM 60
#define SKIPD 10
#define SHDIM 24
#define K2 576          // CDIM + MDIM   (gates GEMM K)
#define G4 2048         // 4*MDIM        (gates GEMM N)
#define FEATD 752       // MDIM + SKIPD*SHDIM
#define ROWS (BDIM*TMDIM)   // 30720 conv-GEMM rows
#define KCONV (KHDIM*NDIM)  // 160

typedef __bf16 bf16_t;
typedef __attribute__((ext_vector_type(16))) __bf16 v16bf;
typedef __attribute__((ext_vector_type(8)))  float  v8f;
typedef __attribute__((ext_vector_type(4)))  unsigned int u32x4;

union BfFrag { v16bf v; u32x4 q[2]; };

// 16-bit A-matrix 16x32 lane layout (ISA 7.12.2): lane m=lane&15, hi=lane>>4.
// Per lane: K = {k0+hi*8 .. +7} then {k0+16+hi*8 .. +7}: two contiguous 16B chunks.
__device__ __forceinline__ v16bf load_a_frag(const bf16_t* p /* rowbase + k0 + hi*8 */) {
  BfFrag f;
  f.q[0] = *(const u32x4*)(p);
  f.q[1] = *(const u32x4*)(p + 16);
  return f.v;
}
// B-matrix 32x16: lane n=lane&15, hi=lane>>4 holds K = {k0+hi*8 .. +7} then {k0+16+hi*8 .. +7}
__device__ __forceinline__ v16bf load_b_frag(const bf16_t* p /* colbase + k0 + hi*8 */) {
  BfFrag f;
  f.q[0] = *(const u32x4*)(p);
  f.q[1] = *(const u32x4*)(p + 16);
  return f.v;
}

__device__ __forceinline__ float sigf(float x) { return 1.0f / (1.0f + __expf(-x)); }

// ---------------- prep kernels ----------------
__global__ void zero_f32(float* p, int n) {
  for (int i = blockIdx.x * blockDim.x + threadIdx.x; i < n; i += gridDim.x * blockDim.x)
    p[i] = 0.0f;
}
__global__ void cvt_bf16(const float* __restrict__ in, bf16_t* __restrict__ outp, int n) {
  for (int i = blockIdx.x * blockDim.x + threadIdx.x; i < n; i += gridDim.x * blockDim.x)
    outp[i] = (bf16_t)in[i];
}
// Wc[n][k] row-major (N=2048 x K=576): k<64 -> W_ih[n][k], else W_hh[n][k-64]
__global__ void build_wc(const float* __restrict__ W_ih, const float* __restrict__ W_hh,
                         bf16_t* __restrict__ Wc) {
  int idx = blockIdx.x * 256 + threadIdx.x;        // exact grid: G4*K2 threads
  int n = idx / K2, k = idx % K2;
  float v = (k < CDIM) ? W_ih[n * CDIM + k] : W_hh[(size_t)n * MDIM + (k - CDIM)];
  Wc[idx] = (bf16_t)v;
}
__global__ void bias_sum(const float* __restrict__ b_ih, const float* __restrict__ b_hh,
                         float* __restrict__ bsum) {
  int i = blockIdx.x * 256 + threadIdx.x;
  if (i < G4) bsum[i] = b_ih[i] + b_hh[i];
}

// ---------------- conv as bf16 WMMA GEMM: (30720 x 160) @ (160 x 64) ----------------
__global__ __launch_bounds__(256) void conv_gemm(const bf16_t* __restrict__ encbf,
                                                 const bf16_t* __restrict__ Wconv,
                                                 const float* __restrict__ convb,
                                                 float* __restrict__ feats) {
  const int lane = threadIdx.x & 31;
  const int wave = threadIdx.x >> 5;
  const int gw = blockIdx.x * 8 + wave;            // 0..1919  (ROWS/16 waves)
  const int mB = gw * 16;
  const int m = lane & 15, hi = lane >> 4;
  const int r = mB + m;
  const int b = r / TMDIM, t = r % TMDIM;
  const bf16_t* ap = encbf + (size_t)b * (TDIM * NDIM) + t * NDIM + hi * 8;  // + k0
  const bf16_t* bp = Wconv + (size_t)m * KCONV + hi * 8;                     // + k0
  v8f a0 = {}, a1 = {}, a2 = {}, a3 = {};
  // register double-buffer: group k+1 loads issued before group-k WMMAs
  v16bf ac  = load_a_frag(ap);
  v16bf bc0 = load_b_frag(bp);
  v16bf bc1 = load_b_frag(bp + 16 * KCONV);
  v16bf bc2 = load_b_frag(bp + 32 * KCONV);
  v16bf bc3 = load_b_frag(bp + 48 * KCONV);
#pragma unroll
  for (int k0 = 0; k0 < KCONV; k0 += 32) {
    v16bf an = ac, bn0 = bc0, bn1 = bc1, bn2 = bc2, bn3 = bc3;
    if (k0 + 32 < KCONV) {
      an  = load_a_frag(ap + k0 + 32);
      bn0 = load_b_frag(bp + k0 + 32);
      bn1 = load_b_frag(bp + k0 + 32 + 16 * KCONV);
      bn2 = load_b_frag(bp + k0 + 32 + 32 * KCONV);
      bn3 = load_b_frag(bp + k0 + 32 + 48 * KCONV);
    }
    a0 = __builtin_amdgcn_wmma_f32_16x16x32_bf16(false, ac, false, bc0, (short)0, a0, false, false);
    a1 = __builtin_amdgcn_wmma_f32_16x16x32_bf16(false, ac, false, bc1, (short)0, a1, false, false);
    a2 = __builtin_amdgcn_wmma_f32_16x16x32_bf16(false, ac, false, bc2, (short)0, a2, false, false);
    a3 = __builtin_amdgcn_wmma_f32_16x16x32_bf16(false, ac, false, bc3, (short)0, a3, false, false);
    ac = an; bc0 = bn0; bc1 = bn1; bc2 = bn2; bc3 = bn3;
  }
  // D layout: vgpr v -> row m = v + 8*hi, col n = lane&15
#pragma unroll
  for (int v = 0; v < 8; ++v) {
    int row = mB + v + 8 * hi;
    float* fr = feats + (size_t)row * CDIM + m;
    fr[0]  = fmaxf(a0[v] + convb[m],       0.0f);
    fr[16] = fmaxf(a1[v] + convb[16 + m],  0.0f);
    fr[32] = fmaxf(a2[v] + convb[32 + m],  0.0f);
    fr[48] = fmaxf(a3[v] + convb[48 + m],  0.0f);
  }
}

// ---------------- Ud[b][c][u] = sum_t feats[b][t][c] * U_e[u][t] ----------------
__global__ __launch_bounds__(256) void ud_kernel(const float* __restrict__ feats,
                                                 const float* __restrict__ U_e,
                                                 float* __restrict__ Ud) {
  __shared__ float fl[TMDIM * CDIM];   // 3840
  __shared__ float ul[TMDIM * TMDIM];  // 3600
  const int b = blockIdx.x, tid = threadIdx.x;
  for (int i = tid; i < TMDIM * CDIM; i += 256) fl[i] = feats[(size_t)b * TMDIM * CDIM + i];
  for (int i = tid; i < TMDIM * TMDIM; i += 256) ul[i] = U_e[i];
  __syncthreads();
  for (int i = tid; i < CDIM * TMDIM; i += 256) {
    int c = i / TMDIM, u = i % TMDIM;
    float a = 0.0f;
    for (int tt = 0; tt < TMDIM; ++tt) a += fl[tt * CDIM + c] * ul[u * TMDIM + tt];
    Ud[(size_t)b * CDIM * TMDIM + i] = a;
  }
}

// ---------------- per-step attention: alpha, writes x_t*alpha (bf16) into abuf[:,0:64] ----------------
__global__ __launch_bounds__(64) void attn_kernel(const float* __restrict__ h,
                                                  const float* __restrict__ s,
                                                  const float* __restrict__ W_e,
                                                  const float* __restrict__ Ud,
                                                  const float* __restrict__ v_e,
                                                  const float* __restrict__ feats,
                                                  bf16_t* __restrict__ abuf, int t) {
  __shared__ float hl[MDIM], sl[MDIM], qwe[TMDIM], sc[CDIM];
  const int b = blockIdx.x, tid = threadIdx.x;
  for (int i = tid; i < MDIM; i += 64) { hl[i] = h[(size_t)b * MDIM + i]; sl[i] = s[(size_t)b * MDIM + i]; }
  __syncthreads();
  if (tid < TMDIM) {
    const float* wr = W_e + (size_t)tid * (2 * MDIM);
    float a = 0.0f;
    for (int j = 0; j < MDIM; ++j) a += hl[j] * wr[j] + sl[j] * wr[MDIM + j];
    qwe[tid] = a;
  }
  __syncthreads();
  {   // tid = channel c (0..63): score = sum_u tanh(qwe[u] + Ud[b,c,u]) * v[u]
    const float* ud = Ud + ((size_t)b * CDIM + tid) * TMDIM;
    float a = 0.0f;
    for (int u = 0; u < TMDIM; ++u) a += tanhf(qwe[u] + ud[u]) * v_e[u];
    sc[tid] = a;
  }
  __syncthreads();
  float mx = -1e30f;
  for (int j = 0; j < CDIM; ++j) mx = fmaxf(mx, sc[j]);
  float sum = 0.0f;
  for (int j = 0; j < CDIM; ++j) sum += __expf(sc[j] - mx);
  float alpha = __expf(sc[tid] - mx) / sum;
  abuf[(size_t)b * K2 + tid] =
      (bf16_t)(feats[((size_t)b * TMDIM + t) * CDIM + tid] * alpha);
}

// ---------------- gates GEMM: (512 x 576) @ (576 x 2048), bf16 WMMA, f32 acc, +bias ----------------
__global__ __launch_bounds__(256) void gates_gemm(const bf16_t* __restrict__ abuf,
                                                  const bf16_t* __restrict__ Wc,
                                                  const float* __restrict__ bsum,
                                                  float* __restrict__ gates) {
  const int lane = threadIdx.x & 31;
  const int wave = threadIdx.x >> 5;
  const int gw = blockIdx.x * 8 + wave;            // 0..1023
  const int ntiles = G4 / 64;                      // 32
  const int mB = (gw / ntiles) * 16;
  const int nB = (gw % ntiles) * 64;
  const int m = lane & 15, hi = lane >> 4;
  const bf16_t* ap = abuf + (size_t)(mB + m) * K2 + hi * 8;          // + k0
  const bf16_t* bp = Wc + (size_t)(nB + m) * K2 + hi * 8;            // + k0
  v8f a0 = {}, a1 = {}, a2 = {}, a3 = {};
  // register double-buffer across K groups
  v16bf ac  = load_a_frag(ap);
  v16bf bc0 = load_b_frag(bp);
  v16bf bc1 = load_b_frag(bp + (size_t)16 * K2);
  v16bf bc2 = load_b_frag(bp + (size_t)32 * K2);
  v16bf bc3 = load_b_frag(bp + (size_t)48 * K2);
#pragma unroll
  for (int k0 = 0; k0 < K2; k0 += 32) {
    v16bf an = ac, bn0 = bc0, bn1 = bc1, bn2 = bc2, bn3 = bc3;
    if (k0 + 32 < K2) {
      an  = load_a_frag(ap + k0 + 32);
      bn0 = load_b_frag(bp + k0 + 32);
      bn1 = load_b_frag(bp + k0 + 32 + (size_t)16 * K2);
      bn2 = load_b_frag(bp + k0 + 32 + (size_t)32 * K2);
      bn3 = load_b_frag(bp + k0 + 32 + (size_t)48 * K2);
    }
    a0 = __builtin_amdgcn_wmma_f32_16x16x32_bf16(false, ac, false, bc0, (short)0, a0, false, false);
    a1 = __builtin_amdgcn_wmma_f32_16x16x32_bf16(false, ac, false, bc1, (short)0, a1, false, false);
    a2 = __builtin_amdgcn_wmma_f32_16x16x32_bf16(false, ac, false, bc2, (short)0, a2, false, false);
    a3 = __builtin_amdgcn_wmma_f32_16x16x32_bf16(false, ac, false, bc3, (short)0, a3, false, false);
    ac = an; bc0 = bn0; bc1 = bn1; bc2 = bn2; bc3 = bn3;
  }
  const float bs0 = bsum[nB + m];
  const float bs1 = bsum[nB + 16 + m];
  const float bs2 = bsum[nB + 32 + m];
  const float bs3 = bsum[nB + 48 + m];
#pragma unroll
  for (int v = 0; v < 8; ++v) {
    int row = mB + v + 8 * hi;
    float* gr = gates + (size_t)row * G4 + nB + m;
    gr[0] = a0[v] + bs0; gr[16] = a1[v] + bs1; gr[32] = a2[v] + bs2; gr[48] = a3[v] + bs3;
  }
}

// ---------------- LSTM cell + beta pooling + bf16 h into abuf[:,64:576] ----------------
__global__ __launch_bounds__(256) void lstm_cell(const float* __restrict__ gates,
                                                 float* __restrict__ h, float* __restrict__ s,
                                                 bf16_t* __restrict__ abuf,
                                                 float* __restrict__ out1acc,
                                                 const float* __restrict__ beta_w, int t) {
  const int idx = blockIdx.x * 256 + threadIdx.x;  // 0..B*M-1
  const int b = idx >> 9, m = idx & (MDIM - 1);
  const float* g = gates + (size_t)b * G4;
  float gi = g[m];
  float gf = g[MDIM + m];
  float gg = g[2 * MDIM + m];
  float go = g[3 * MDIM + m];
  float c  = sigf(gf) * s[idx] + sigf(gi) * tanhf(gg);
  float hn = sigf(go) * tanhf(c);
  s[idx] = c;
  h[idx] = hn;
  abuf[(size_t)b * K2 + CDIM + m] = (bf16_t)hn;
  out1acc[idx] += beta_w[t] * hn;
}

// ---------------- out1 -> featbuf[:, 0:512] ----------------
__global__ void out1_kernel(const float* __restrict__ out1acc, const float* __restrict__ beta_b,
                            float* __restrict__ featbuf) {
  const int idx = blockIdx.x * 256 + threadIdx.x;
  const int b = idx >> 9, m = idx & (MDIM - 1);
  featbuf[(size_t)b * FEATD + m] = fmaxf(out1acc[idx] + beta_b[0], 0.0f);
}

// ---------------- GRU skip branch: 5120 independent rows, 6 steps ----------------
__global__ __launch_bounds__(96) void gru_kernel(const float* __restrict__ feats,
                                                 const float* __restrict__ gW_ih,
                                                 const float* __restrict__ gW_hh,
                                                 const float* __restrict__ gb_ih,
                                                 const float* __restrict__ gb_hh,
                                                 float* __restrict__ featbuf) {
  __shared__ float xg[CDIM], gi[3 * SHDIM], gh[3 * SHDIM], hb[SHDIM];
  const int row = blockIdx.x;              // 0..5119
  const int b = row / SKIPD, j = row % SKIPD;
  const int tid = threadIdx.x;
  if (tid < SHDIM) hb[tid] = 0.0f;
  __syncthreads();
  for (int p = 0; p < TMDIM / SKIPD; ++p) {
    if (tid < CDIM) xg[tid] = feats[((size_t)b * TMDIM + p * SKIPD + j) * CDIM + tid];
    __syncthreads();
    if (tid < 3 * SHDIM) {
      const float* wi = gW_ih + (size_t)tid * CDIM;
      float a = gb_ih[tid];
      for (int c = 0; c < CDIM; ++c) a += xg[c] * wi[c];
      gi[tid] = a;
      const float* wh = gW_hh + (size_t)tid * SHDIM;
      float hsum = gb_hh[tid];
      for (int u = 0; u < SHDIM; ++u) hsum += hb[u] * wh[u];
      gh[tid] = hsum;
    }
    __syncthreads();
    if (tid < SHDIM) {
      float r  = sigf(gi[tid] + gh[tid]);
      float z  = sigf(gi[SHDIM + tid] + gh[SHDIM + tid]);
      float nn = tanhf(gi[2 * SHDIM + tid] + r * gh[2 * SHDIM + tid]);
      hb[tid] = (1.0f - z) * nn + z * hb[tid];
    }
    __syncthreads();
  }
  if (tid < SHDIM) featbuf[(size_t)b * FEATD + MDIM + j * SHDIM + tid] = hb[tid];
}

// ---------------- final MLP: y[b] = relu(feat @ W1^T + b1) @ W2^T + b2 ----------------
__global__ __launch_bounds__(256) void final_kernel(const float* __restrict__ featbuf,
                                                    const float* __restrict__ W1,
                                                    const float* __restrict__ b1,
                                                    const float* __restrict__ W2,
                                                    const float* __restrict__ b2,
                                                    float* __restrict__ out) {
  __shared__ float f[FEATD];
  __shared__ float red[256];
  const int b = blockIdx.x, tid = threadIdx.x;
  for (int i = tid; i < FEATD; i += 256) f[i] = featbuf[(size_t)b * FEATD + i];
  __syncthreads();
  float acc = 0.0f;
  for (int m = tid; m < MDIM; m += 256) {
    const float* w = W1 + (size_t)m * FEATD;
    float d = b1[m];
    for (int k = 0; k < FEATD; ++k) d += f[k] * w[k];
    acc += fmaxf(d, 0.0f) * W2[m];
  }
  red[tid] = acc;
  __syncthreads();
  for (int off2 = 128; off2 > 0; off2 >>= 1) {
    if (tid < off2) red[tid] += red[tid + off2];
    __syncthreads();
  }
  if (tid == 0) out[b] = red[0] + b2[0];
}

extern "C" void kernel_launch(void* const* d_in, const int* in_sizes, int n_in,
                              void* d_out, int out_size, void* d_ws, size_t ws_size,
                              hipStream_t stream) {
  (void)in_sizes; (void)n_in; (void)out_size; (void)ws_size;
  const float* enc    = (const float*)d_in[0];
  const float* convw  = (const float*)d_in[1];
  const float* convb  = (const float*)d_in[2];
  const float* W_e    = (const float*)d_in[3];
  const float* U_e    = (const float*)d_in[4];
  const float* v_e    = (const float*)d_in[5];
  const float* W_ih   = (const float*)d_in[6];
  const float* W_hh   = (const float*)d_in[7];
  const float* b_ih   = (const float*)d_in[8];
  const float* b_hh   = (const float*)d_in[9];
  const float* beta_w = (const float*)d_in[10];
  const float* beta_b = (const float*)d_in[11];
  const float* gW_ih  = (const float*)d_in[12];
  const float* gW_hh  = (const float*)d_in[13];
  const float* gb_ih  = (const float*)d_in[14];
  const float* gb_hh  = (const float*)d_in[15];
  const float* W1     = (const float*)d_in[16];
  const float* b1     = (const float*)d_in[17];
  const float* W2     = (const float*)d_in[18];
  const float* b2     = (const float*)d_in[19];
  float* out = (float*)d_out;

  char* ws = (char*)d_ws;
  size_t off = 0;
  auto take = [&](size_t bytes) -> char* {
    char* p = ws + off;
    off += (bytes + 255) & ~(size_t)255;
    return p;
  };
  float*  feats   = (float*)take((size_t)ROWS * CDIM * 4);            // 7.9 MB
  float*  Ud      = (float*)take((size_t)BDIM * CDIM * TMDIM * 4);    // 7.9 MB
  float*  hbuf    = (float*)take((size_t)BDIM * MDIM * 4);
  float*  sbuf    = (float*)take((size_t)BDIM * MDIM * 4);
  float*  out1acc = (float*)take((size_t)BDIM * MDIM * 4);
  float*  gates   = (float*)take((size_t)BDIM * G4 * 4);              // 4 MB
  bf16_t* abuf    = (bf16_t*)take((size_t)BDIM * K2 * 2);
  bf16_t* Wc      = (bf16_t*)take((size_t)G4 * K2 * 2);               // 2.4 MB
  float*  bsum    = (float*)take((size_t)G4 * 4);
  bf16_t* encbf   = (bf16_t*)take((size_t)BDIM * TDIM * NDIM * 2);    // 2 MB
  bf16_t* Wconvbf = (bf16_t*)take((size_t)CDIM * KCONV * 2);
  float*  featbuf = (float*)take((size_t)BDIM * FEATD * 4);           // 1.5 MB

  // ---- prep ----
  zero_f32<<<512, 256, 0, stream>>>(hbuf, BDIM * MDIM);
  zero_f32<<<512, 256, 0, stream>>>(sbuf, BDIM * MDIM);
  zero_f32<<<512, 256, 0, stream>>>(out1acc, BDIM * MDIM);
  zero_f32<<<512, 256, 0, stream>>>((float*)abuf, BDIM * K2 / 2);     // zero bits = bf16 zero
  cvt_bf16<<<1024, 256, 0, stream>>>(enc, encbf, BDIM * TDIM * NDIM);
  cvt_bf16<<<40, 256, 0, stream>>>(convw, Wconvbf, CDIM * KCONV);
  build_wc<<<(G4 * K2) / 256, 256, 0, stream>>>(W_ih, W_hh, Wc);
  bias_sum<<<8, 256, 0, stream>>>(b_ih, b_hh, bsum);

  // ---- feed-forward features ----
  conv_gemm<<<ROWS / 16 / 8, 256, 0, stream>>>(encbf, Wconvbf, convb, feats);  // 240 blocks
  ud_kernel<<<BDIM, 256, 0, stream>>>(feats, U_e, Ud);

  // ---- attention-LSTM scan (sequential in t) ----
  for (int t = 0; t < TMDIM; ++t) {
    attn_kernel<<<BDIM, 64, 0, stream>>>(hbuf, sbuf, W_e, Ud, v_e, feats, abuf, t);
    gates_gemm<<<(BDIM / 16) * (G4 / 64) / 8, 256, 0, stream>>>(abuf, Wc, bsum, gates); // 128 blocks
    lstm_cell<<<(BDIM * MDIM) / 256, 256, 0, stream>>>(gates, hbuf, sbuf, abuf,
                                                       out1acc, beta_w, t);
  }

  // ---- heads ----
  out1_kernel<<<(BDIM * MDIM) / 256, 256, 0, stream>>>(out1acc, beta_b, featbuf);
  gru_kernel<<<BDIM * SKIPD, 96, 0, stream>>>(feats, gW_ih, gW_hh, gb_ih, gb_hh, featbuf);
  final_kernel<<<BDIM, 256, 0, stream>>>(featbuf, W1, b1, W2, b2, out);
}
